// ComplexAttention_27058293965608
// MI455X (gfx1250) — matrix operand order repaired and checked by reading, not compiled, not run on hardware
//
#include <hip/hip_runtime.h>

// ---------------------------------------------------------------------------
// Complex attention on gfx1250 via V_WMMA_F32_16X16X32_F16.
// B=4, L=S=1024, H=8, E=D=64.
// ---------------------------------------------------------------------------

typedef _Float16 h16;
typedef __attribute__((ext_vector_type(8)))  _Float16 v8h_t;
typedef __attribute__((ext_vector_type(16))) _Float16 v16h_t;
typedef __attribute__((ext_vector_type(8)))  float    v8f_t;

#define WMMA_F16(a, b, c) \
    __builtin_amdgcn_wmma_f32_16x16x32_f16(false, (a), false, (b), (short)0, (c), false, false)

// Load 16 f16 from two (possibly disjoint) 16B-aligned 8-element runs.
static __device__ __forceinline__ v16h_t ld16(const h16* p0, const h16* p1) {
    v8h_t a = *(const v8h_t*)p0;
    v8h_t b = *(const v8h_t*)p1;
    return __builtin_shufflevector(a, b, 0, 1, 2, 3, 4, 5, 6, 7,
                                         8, 9, 10, 11, 12, 13, 14, 15);
}

// ---- fast exp2 ------------------------------------------------------------
static __device__ __forceinline__ float fast_exp2(float x) {
#if __has_builtin(__builtin_amdgcn_exp2f)
    return __builtin_amdgcn_exp2f(x);
#else
    return exp2f(x);
#endif
}

// ---- intra-row (16-lane) xor butterfly reductions via v_permlane16_b32 ----
#if __has_builtin(__builtin_amdgcn_permlane16)
static __device__ __forceinline__ float pl16(float v, unsigned s0, unsigned s1) {
    return __int_as_float(__builtin_amdgcn_permlane16(
        __float_as_int(v), __float_as_int(v), s0, s1, false, false));
}
static __device__ __forceinline__ float rowmax16(float v) {
    v = fmaxf(v, pl16(v, 0x67452301u, 0xEFCDAB89u)); // xor 1
    v = fmaxf(v, pl16(v, 0x54761032u, 0xDCFE98BAu)); // xor 2
    v = fmaxf(v, pl16(v, 0x32107654u, 0xBA98FEDCu)); // xor 4
    v = fmaxf(v, pl16(v, 0xFEDCBA98u, 0x76543210u)); // xor 8
    return v;
}
static __device__ __forceinline__ float rowsum16(float v) {
    v += pl16(v, 0x67452301u, 0xEFCDAB89u);
    v += pl16(v, 0x54761032u, 0xDCFE98BAu);
    v += pl16(v, 0x32107654u, 0xBA98FEDCu);
    v += pl16(v, 0xFEDCBA98u, 0x76543210u);
    return v;
}
#else
static __device__ __forceinline__ float rowmax16(float v) {
    for (int off = 8; off >= 1; off >>= 1) v = fmaxf(v, __shfl_xor(v, off, 32));
    return v;
}
static __device__ __forceinline__ float rowsum16(float v) {
    for (int off = 8; off >= 1; off >>= 1) v += __shfl_xor(v, off, 32);
    return v;
}
#endif

// ---------------------------------------------------------------------------
// Pass 1: f32 -> f16 pack with layout change.
//   Q,K:  [b,l|s,h,e] -> [b,h,l|s,e]     (row-major per (b,h))
//   V:    [b,s,h,d]   -> [b,h,d,s]       (transposed: d-major rows)
// N = B*H*1024*64 = 2,097,152 elements per array.
// ---------------------------------------------------------------------------
__global__ __launch_bounds__(256) void pack_f16(
    const float* __restrict__ qr, const float* __restrict__ qi,
    const float* __restrict__ kr, const float* __restrict__ ki,
    const float* __restrict__ vr, const float* __restrict__ vi,
    h16* __restrict__ oQr, h16* __restrict__ oQi,
    h16* __restrict__ oKr, h16* __restrict__ oKi,
    h16* __restrict__ oVr, h16* __restrict__ oVi)
{
    const int t = blockIdx.x * 256 + threadIdx.x;      // output-flat index
    // Q/K: out [b][h][l][e]
    const int e = t & 63;
    const int l = (t >> 6) & 1023;
    const int h = (t >> 16) & 7;
    const int b = t >> 19;
    const int qin = ((b * 1024 + l) * 8 + h) * 64 + e; // in [b][l][h][e]
    oQr[t] = (h16)qr[qin];
    oQi[t] = (h16)qi[qin];
    oKr[t] = (h16)kr[qin];
    oKi[t] = (h16)ki[qin];
    // V: out [b][h][d][s]
    const int s = t & 1023;
    const int d = (t >> 10) & 63;
    const int vin = ((b * 1024 + s) * 8 + h) * 64 + d; // in [b][s][h][d]
    oVr[t] = (h16)vr[vin];
    oVi[t] = (h16)vi[vin];
}

// ---------------------------------------------------------------------------
// Pass 2: flash-style complex attention. One wave per 16-row L tile.
// ---------------------------------------------------------------------------
__global__ __launch_bounds__(256, 1) void cattn_wmma(
    const h16* __restrict__ Qr, const h16* __restrict__ Qi,
    const h16* __restrict__ Kr, const h16* __restrict__ Ki,
    const h16* __restrict__ Vr, const h16* __restrict__ Vi,
    float* __restrict__ out)
{
    // per-wave P staging: [wave][channel][16 rows][32 cols padded to 40]
    __shared__ _Float16 Pl[8][2][16][40];

    const int tid  = threadIdx.x;
    const int wave = tid >> 5;
    const int lane = tid & 31;
    const int half = lane >> 4;   // 0: lanes 0-15, 1: lanes 16-31
    const int r    = lane & 15;

    const int gw = blockIdx.x * 8 + wave;  // 0..2047
    const int lt = gw & 63;                // L tile (16 rows each)
    const int bh = gw >> 6;                // b*H + h

    const size_t matB = (size_t)1024 * 64; // per-(b,h) matrix size
    const h16* qr_p = Qr + (size_t)bh * matB + (size_t)(lt * 16 + r) * 64;
    const h16* qi_p = Qi + (size_t)bh * matB + (size_t)(lt * 16 + r) * 64;
    const h16* kr_b = Kr + (size_t)bh * matB;
    const h16* ki_b = Ki + (size_t)bh * matB;
    const h16* vr_b = Vr + (size_t)bh * matB;
    const h16* vi_b = Vi + (size_t)bh * matB;

    // ---- Q fragments (A-layout, 16x32 per E-chunk), resident for whole kernel
    v16h_t qrF[2], qiF[2], qrN[2];
#pragma unroll
    for (int c = 0; c < 2; ++c) {
        const h16* p0 = qr_p + c * 32 + half * 8;
        qrF[c] = ld16(p0, p0 + 16);
        const h16* p1 = qi_p + c * 32 + half * 8;
        qiF[c] = ld16(p1, p1 + 16);
        qrN[c] = -qrF[c];   // FP16 WMMA has no A-negate; negate in registers
    }

    // ---- accumulators: four product channels, 4 d-tiles of 16 each
    v8f_t accRR[4] = {}, accRI[4] = {}, accIR[4] = {}, accII[4] = {};
    float m_s[2][8], l_s[2][8];
#pragma unroll
    for (int ch = 0; ch < 2; ++ch)
#pragma unroll
        for (int j = 0; j < 8; ++j) { m_s[ch][j] = -1e30f; l_s[ch][j] = 0.0f; }

    // fold softmax scale and log2(e) into the scores once (exp2 domain)
    const float c2 = 0.125f * 1.44269504088896340736f; // (1/sqrt(E)) * log2(e)

    for (int s0 = 0; s0 < 1024; s0 += 32) {
        // ---------------- scores: Sr = QrKr^T + QiKi^T ; Si = QiKr^T - QrKi^T
        v8f_t S[2][2]; // [channel][16-col subtile]
#pragma unroll
        for (int t = 0; t < 2; ++t) {
            const int sc = s0 + t * 16 + r;           // this lane's K row (s)
            const h16* kp0 = kr_b + (size_t)sc * 64 + half * 8;
            const h16* kp1 = kp0 + 32;
            const h16* ip0 = ki_b + (size_t)sc * 64 + half * 8;
            const h16* ip1 = ip0 + 32;
            v16h_t kr0 = ld16(kp0, kp0 + 16);
            v16h_t kr1 = ld16(kp1, kp1 + 16);
            v16h_t ki0 = ld16(ip0, ip0 + 16);
            v16h_t ki1 = ld16(ip1, ip1 + 16);
            v8f_t z = {};
            v8f_t sr = WMMA_F16(qrF[0], kr0, z);
            sr = WMMA_F16(qrF[1], kr1, sr);
            sr = WMMA_F16(qiF[0], ki0, sr);
            sr = WMMA_F16(qiF[1], ki1, sr);
            v8f_t si = WMMA_F16(qiF[0], kr0, z);
            si = WMMA_F16(qiF[1], kr1, si);
            si = WMMA_F16(qrN[0], ki0, si);
            si = WMMA_F16(qrN[1], ki1, si);
            S[0][t] = sr;
            S[1][t] = si;
        }

        // ---------------- online softmax per channel (half-wave reductions)
#pragma unroll
        for (int ch = 0; ch < 2; ++ch) {
            float mloc[8], corr[8], rs[8];
#pragma unroll
            for (int j = 0; j < 8; ++j) {
                // scale into exp2 domain once
                S[ch][0][j] *= c2;
                S[ch][1][j] *= c2;
                mloc[j] = fmaxf(S[ch][0][j], S[ch][1][j]);
            }
#pragma unroll
            for (int j = 0; j < 8; ++j) mloc[j] = rowmax16(mloc[j]);
#pragma unroll
            for (int j = 0; j < 8; ++j) {
                const float mn = fmaxf(m_s[ch][j], mloc[j]);
                corr[j]    = fast_exp2(m_s[ch][j] - mn);
                m_s[ch][j] = mn;
                const float a = fast_exp2(S[ch][0][j] - mn);
                const float b = fast_exp2(S[ch][1][j] - mn);
                S[ch][0][j] = a;
                S[ch][1][j] = b;
                rs[j] = a + b;
            }
#pragma unroll
            for (int j = 0; j < 8; ++j) rs[j] = rowsum16(rs[j]);
#pragma unroll
            for (int j = 0; j < 8; ++j)
                l_s[ch][j] = l_s[ch][j] * corr[j] + rs[j];

            // rescale the two accumulators fed by this channel's P
            v8f_t* A0 = ch ? accIR : accRR;
            v8f_t* A1 = ch ? accII : accRI;
#pragma unroll
            for (int d = 0; d < 4; ++d)
#pragma unroll
                for (int j = 0; j < 8; ++j) {
                    A0[d][j] *= corr[j];
                    A1[d][j] *= corr[j];
                }

            // stash P (f16) into LDS in row-major [l][s] for A-layout reload
#pragma unroll
            for (int j = 0; j < 8; ++j) {
                Pl[wave][ch][half * 8 + j][r]      = (_Float16)S[ch][0][j];
                Pl[wave][ch][half * 8 + j][16 + r] = (_Float16)S[ch][1][j];
            }
        }
        asm volatile("" ::: "memory"); // keep DS store->load order (same wave: HW in-order)

        // ---------------- reload P as 16x32 A-operands
        const _Float16* ppr = &Pl[wave][0][r][half * 8];
        v16h_t Pr = ld16(ppr, ppr + 16);
        const _Float16* ppi = &Pl[wave][1][r][half * 8];
        v16h_t Pi = ld16(ppi, ppi + 16);
        asm volatile("" ::: "memory");

        // ---------------- A·V over this s-chunk, 4 d-tiles
#pragma unroll
        for (int d = 0; d < 4; ++d) {
            const h16* vpr = vr_b + (size_t)(d * 16 + r) * 1024 + s0 + half * 8;
            const h16* vpi = vi_b + (size_t)(d * 16 + r) * 1024 + s0 + half * 8;
            v16h_t Vrf = ld16(vpr, vpr + 16);
            v16h_t Vif = ld16(vpi, vpi + 16);
            accRR[d] = WMMA_F16(Pr, Vrf, accRR[d]);
            accRI[d] = WMMA_F16(Pr, Vif, accRI[d]);
            accIR[d] = WMMA_F16(Pi, Vrf, accIR[d]);
            accII[d] = WMMA_F16(Pi, Vif, accII[d]);
        }
    }

    // ---------------- epilogue: normalize, combine complex parts, store
    float ilr[8], ili[8];
#pragma unroll
    for (int j = 0; j < 8; ++j) {
        ilr[j] = 1.0f / l_s[0][j];
        ili[j] = 1.0f / l_s[1][j];
    }
    const int b = bh >> 3;
    const int h = bh & 7;
#pragma unroll
    for (int d = 0; d < 4; ++d) {
#pragma unroll
        for (int j = 0; j < 8; ++j) {
            const int lg = lt * 16 + half * 8 + j;  // global l row
            const int dg = d * 16 + r;              // global d col
            const size_t o_r = ((((size_t)0 * 4 + b) * 1024 + lg) * 8 + h) * 64 + dg;
            const size_t o_i = ((((size_t)1 * 4 + b) * 1024 + lg) * 8 + h) * 64 + dg;
            out[o_r] = accRR[d][j] * ilr[j] - accII[d][j] * ili[j];
            out[o_i] = accRI[d][j] * ilr[j] + accIR[d][j] * ili[j];
        }
    }
}

// ---------------------------------------------------------------------------
extern "C" void kernel_launch(void* const* d_in, const int* in_sizes, int n_in,
                              void* d_out, int out_size, void* d_ws, size_t ws_size,
                              hipStream_t stream) {
    const float* qr = (const float*)d_in[0];
    const float* qi = (const float*)d_in[1];
    const float* kr = (const float*)d_in[2];
    const float* ki = (const float*)d_in[3];
    const float* vr = (const float*)d_in[4];
    const float* vi = (const float*)d_in[5];
    // d_in[6] = attn_mask (unused, mask_flag=False)

    const size_t N = (size_t)4 * 8 * 1024 * 64; // elems per packed array
    h16* ws  = (h16*)d_ws;
    h16* Qr = ws;
    h16* Qi = ws + N;
    h16* Kr = ws + 2 * N;
    h16* Ki = ws + 3 * N;
    h16* Vt = ws + 4 * N;   // V real transposed [b][h][d][s]
    h16* Vu = ws + 5 * N;   // V imag transposed

    pack_f16<<<(int)(N / 256), 256, 0, stream>>>(qr, qi, kr, ki, vr, vi,
                                                 Qr, Qi, Kr, Ki, Vt, Vu);
    // 2048 waves total: 8 waves (256 thr) per block, 256 blocks
    cattn_wmma<<<256, 256, 0, stream>>>(Qr, Qi, Kr, Ki, Vt, Vu, (float*)d_out);
}
